// MotionEncoder_4810363372447
// MI455X (gfx1250) — matrix operand order repaired and checked by reading, not compiled, not run on hardware
//
#include <hip/hip_runtime.h>

typedef __attribute__((ext_vector_type(16))) _Float16 v16h;
typedef __attribute__((ext_vector_type(8)))  _Float16 v8h;
typedef __attribute__((ext_vector_type(8)))  float    v8f;
typedef int v4i __attribute__((__vector_size__(16)));   // matches builtin param type

#if defined(__has_builtin)
#if __has_builtin(__builtin_amdgcn_global_load_async_to_lds_b128) && \
    __has_builtin(__builtin_amdgcn_s_wait_asynccnt)
#define HAVE_ASYNC_LDS 1
#endif
#endif
#ifndef HAVE_ASYNC_LDS
#define HAVE_ASYNC_LDS 0
#endif

// ---------------------------------------------------------------------------
// Weight pre-pack: raw fp32 [O][Cin][kx][ky][kz][kt] -> f16 WMMA B-fragments
// wp[((chunk*tilesN + tn)*32 + lane)*16 + h] ; zero-padded beyond Cout/Ktot
// ---------------------------------------------------------------------------
struct PackDesc { int Cin, Cout, taps; };

__global__ void pack_w_kernel(const float* __restrict__ w, _Float16* __restrict__ wp,
                              PackDesc d, int nchunks) {
  const int tilesN = (d.Cout + 15) >> 4;
  int i = blockIdx.x * blockDim.x + threadIdx.x;
  if (i >= nchunks * tilesN * 32) return;
  const int lane  = i & 31;
  const int tn    = (i >> 5) % tilesN;
  const int chunk = (i >> 5) / tilesN;
  const int col   = tn * 16 + (lane & 15);
  const int hi    = lane >> 4;
  const int Ktot  = d.taps * d.Cin;
#pragma unroll
  for (int h = 0; h < 16; ++h) {
    const int k = chunk * 32 + (h & 7) + (hi << 3) + ((h >> 3) << 4);
    float val = 0.f;
    if (col < d.Cout && k < Ktot) {
      const int tap = k / d.Cin;
      const int ci  = k - tap * d.Cin;
      val = w[(col * d.Cin + ci) * d.taps + tap];
    }
    wp[(size_t)i * 16 + h] = (_Float16)val;
  }
}

// ---------------------------------------------------------------------------
// Specialized implicit-GEMM conv: f16 activations in, f32 out.
// Wave tile: 16 voxels x NT*16 out-channels.
// FASTA (CIN%32==0): one tap per chunk -> 2 vector loads per lane for A.
// STAGE (TILESM%8==0): B fragments staged block-wide into LDS (async when
// available), double-buffered; else direct per-wave global B loads.
// ---------------------------------------------------------------------------
template <int SI, int SO, int CIN, int COUT, int K, int STRIDE, int PX, int PT,
          int MODE, int NT>
__global__ __launch_bounds__(256)
void conv_wmma_t(const _Float16* __restrict__ in, const _Float16* __restrict__ wp,
                 float* __restrict__ out)
{
  constexpr int  NVOX   = SO * SO * SO * 4;
  constexpr int  TAPS   = K * K * K * K;
  constexpr int  KTOT   = TAPS * CIN;
  constexpr int  NCHUNK = (KTOT + 31) / 32;
  constexpr int  TILESM = NVOX / 16;
  constexpr int  TILESN = (COUT + 15) / 16;
  constexpr int  NGROUP = TILESN / NT;
  constexpr bool FASTA  = (CIN % 32) == 0;
  constexpr bool STAGE  = (TILESM % 8) == 0;
  static_assert(NVOX % 16 == 0, "");
  static_assert(TILESN % NT == 0, "");
  static_assert(!FASTA || (KTOT % 32) == 0, "");

  constexpr int SBN = STAGE ? 2 * 2048 : 32;       // halves (two 4KB buffers)
  __shared__ __align__(32) _Float16 sB[SBN];

  const int tid  = threadIdx.x;
  const int lane = tid & 31;
  const int wave = blockIdx.x * 8 + (tid >> 5);
  if (!STAGE && wave >= TILESM * NGROUP) return;
  const int tm = wave % TILESM;
  const int ng = wave / TILESM;                    // uniform per block if STAGE
  const int hi = lane >> 4;

  int v = tm * 16 + (lane & 15);
  const int to = v & 3;  v >>= 2;
  const int zo = v % SO; v /= SO;
  const int yo = v % SO; v /= SO;
  const int xo = v;

  v8f acc[NT];
#pragma unroll
  for (int t = 0; t < NT; ++t) acc[t] = (v8f){};

  // ---- A fragment builder for chunk c ----
  auto buildA = [&](int c) -> v16h {
    union { v16h v; v8h h8[2]; _Float16 e[16]; } av;
    if constexpr (FASTA) {
      constexpr int CQ = CIN / 32;
      const int tap = c / CQ;
      const int cib = (c - tap * CQ) * 32;
      const int at = tap % K;
      const int az = (tap / K) % K;
      const int ay = (tap / (K * K)) % K;
      const int ax = tap / (K * K * K);
      int xi, yi, zi, ti;
      bool ok = true;
      if (MODE == 0) {
        xi = xo * STRIDE + ax - PX;
        yi = yo * STRIDE + ay - PX;
        zi = zo * STRIDE + az - PX;
        ti = to + at - PT;
      } else {
        xi = xo - ax; ok = ok && ((xi & 1) == 0) && (xi >= 0); xi >>= 1;
        yi = yo - ay; ok = ok && ((yi & 1) == 0) && (yi >= 0); yi >>= 1;
        zi = zo - az; ok = ok && ((zi & 1) == 0) && (zi >= 0); zi >>= 1;
        ti = to - at;
      }
      ok = ok && (xi >= 0) && (xi < SI) && (yi >= 0) && (yi < SI) &&
                 (zi >= 0) && (zi < SI) && (ti >= 0) && (ti < 4);
      av.h8[0] = (v8h){};
      av.h8[1] = (v8h){};
      if (ok) {
        const int vi = ((xi * SI + yi) * SI + zi) * 4 + ti;
        const _Float16* p = in + (size_t)vi * CIN + cib + (hi << 3);
        av.h8[0] = *(const v8h*)p;          // halves 0-7  : K local (0..7)+8hi
        av.h8[1] = *(const v8h*)(p + 16);   // halves 8-15 : K local (16..23)+8hi
      }
    } else {
      const int k0 = c * 32;
#pragma unroll
      for (int h = 0; h < 16; ++h) {
        const int j  = h >> 1;
        const int kl = ((j < 4) ? (2 * j) : (16 + 2 * (j - 4))) + (h & 1) + (hi << 3);
        const int k  = k0 + kl;
        _Float16 val = (_Float16)0.f;
        if (k < KTOT) {
          const int tap = k / CIN;
          const int ci  = k - tap * CIN;
          const int at  = tap % K;
          const int az  = (tap / K) % K;
          const int ay  = (tap / (K * K)) % K;
          const int ax  = tap / (K * K * K);
          int xi, yi, zi, ti;
          bool ok = true;
          if (MODE == 0) {
            xi = xo * STRIDE + ax - PX;
            yi = yo * STRIDE + ay - PX;
            zi = zo * STRIDE + az - PX;
            ti = to + at - PT;
          } else {
            xi = xo - ax; ok = ok && ((xi & 1) == 0) && (xi >= 0); xi >>= 1;
            yi = yo - ay; ok = ok && ((yi & 1) == 0) && (yi >= 0); yi >>= 1;
            zi = zo - az; ok = ok && ((zi & 1) == 0) && (zi >= 0); zi >>= 1;
            ti = to - at;
          }
          ok = ok && (xi >= 0) && (xi < SI) && (yi >= 0) && (yi < SI) &&
                     (zi >= 0) && (zi < SI) && (ti >= 0) && (ti < 4);
          if (ok) {
            const int vi = ((xi * SI + yi) * SI + zi) * 4 + ti;
            val = in[(size_t)vi * CIN + ci];
          }
        }
        av.e[h] = val;
      }
    }
    return av.v;
  };

  if constexpr (STAGE) {
    // Block-wide staged B (all 8 waves share ng): double-buffered LDS.
    auto prefetch = [&](int buf, int c) {
      if (tid < NT * 64) {
        const _Float16* g = wp + ((size_t)c * TILESN + ng * NT) * 512 + tid * 8;
        _Float16* l = &sB[buf * 2048 + tid * 8];
#if HAVE_ASYNC_LDS
        __builtin_amdgcn_global_load_async_to_lds_b128(
            (__attribute__((address_space(1))) v4i*)(v4i*)g,
            (__attribute__((address_space(3))) v4i*)(v4i*)l, 0, 0);
#else
        *(v4i*)l = *(const v4i*)g;
#endif
      }
    };
    auto waitAsync = [&]() {
#if HAVE_ASYNC_LDS
      __builtin_amdgcn_s_wait_asynccnt(0);
#endif
    };

    int buf = 0;
    prefetch(0, 0);
    waitAsync();
    __syncthreads();
    for (int c = 0; c < NCHUNK; ++c) {
      if (c + 1 < NCHUNK) prefetch(buf ^ 1, c + 1);   // overlaps with consume
      const v16h a = buildA(c);
#pragma unroll
      for (int t = 0; t < NT; ++t) {
        const v16h b = *(const v16h*)&sB[buf * 2048 + (t * 32 + lane) * 16];
        acc[t] = __builtin_amdgcn_wmma_f32_16x16x32_f16(
            false, a, false, b, (short)0, acc[t], false, false);
      }
      __syncthreads();                                 // consumers done with buf
      if (c + 1 < NCHUNK) { waitAsync(); __syncthreads(); }  // next chunk ready
      buf ^= 1;
    }
  } else {
    // Direct per-wave B loads (tiny S=2 layers).
    for (int c = 0; c < NCHUNK; ++c) {
      const v16h a = buildA(c);
      const v16h* bp = (const v16h*)wp + ((size_t)c * TILESN + ng * NT) * 32 + lane;
#pragma unroll
      for (int t = 0; t < NT; ++t) {
        const v16h b = bp[t * 32];
        acc[t] = __builtin_amdgcn_wmma_f32_16x16x32_f16(
            false, a, false, b, (short)0, acc[t], false, false);
      }
    }
  }

  // D store: VGPR j -> M = j + 8*hi, N = lane&15
#pragma unroll
  for (int t = 0; t < NT; ++t) {
    const int tn = ng * NT + t;
#pragma unroll
    for (int j = 0; j < 8; ++j) {
      const int m  = j + (hi << 3);
      const int vv = tm * 16 + m;
      const int n  = tn * 16 + (lane & 15);
      if (COUT % 16 == 0 || n < COUT) out[vv * COUT + n] = acc[t][j];
    }
  }
}

// ---------------------------------------------------------------------------
// Support kernels (activations are f16; conv outputs / BN stats are f32)
// ---------------------------------------------------------------------------
__global__ void zero_h_kernel(_Float16* __restrict__ p, int n) {
  int i = blockIdx.x * blockDim.x + threadIdx.x;
  if (i < n) p[i] = (_Float16)0.f;
}

__global__ void voxelize_kernel(const float* __restrict__ pcd, _Float16* __restrict__ grid, int npts) {
  int i = blockIdx.x * blockDim.x + threadIdx.x;
  if (i >= npts) return;
  int ix = (int)floorf(pcd[i * 4 + 0] / 0.1f);
  int iy = (int)floorf(pcd[i * 4 + 1] / 0.1f);
  int iz = (int)floorf(pcd[i * 4 + 2] / 0.1f);
  int it = (int)floorf(pcd[i * 4 + 3] / 1.0f);
  if (ix < 0 || ix >= 32 || iy < 0 || iy >= 32 || iz < 0 || iz >= 32 || it < 0 || it >= 4) return;
  grid[((ix * 32 + iy) * 32 + iz) * 4 + it] = (_Float16)0.5f;
}

__global__ void bn_stats_kernel(const float* __restrict__ x, float* __restrict__ stats,
                                int Nvox, int C) {
  const int c = blockIdx.x;
  float s = 0.f, s2 = 0.f;
  for (int v = blockIdx.y * blockDim.x + threadIdx.x; v < Nvox; v += gridDim.y * blockDim.x) {
    float t = x[v * C + c];
    s += t; s2 += t * t;
  }
  __shared__ float ls[256], ls2[256];
  ls[threadIdx.x] = s; ls2[threadIdx.x] = s2;
  __syncthreads();
  for (int o = 128; o > 0; o >>= 1) {
    if (threadIdx.x < o) { ls[threadIdx.x] += ls[threadIdx.x + o]; ls2[threadIdx.x] += ls2[threadIdx.x + o]; }
    __syncthreads();
  }
  if (threadIdx.x == 0) {
    atomicAdd(&stats[c], ls[0]);
    atomicAdd(&stats[C + c], ls2[0]);
  }
}

__global__ void zero_kernel(float* __restrict__ p, int n) {
  int i = blockIdx.x * blockDim.x + threadIdx.x;
  if (i < n) p[i] = 0.f;
}

__global__ void bn_finalize_kernel(float* __restrict__ stats, int Nvox, int C) {
  int c = blockIdx.x * blockDim.x + threadIdx.x;
  if (c >= C) return;
  float m = stats[c] / (float)Nvox;
  float v = stats[C + c] / (float)Nvox - m * m;
  stats[2 * C + c] = m;
  stats[3 * C + c] = rsqrtf(v + 1e-5f);
}

__global__ void bn_apply_kernel(const float* __restrict__ x, const float* __restrict__ stats,
                                const float* __restrict__ g, const float* __restrict__ b,
                                const _Float16* __restrict__ res, _Float16* __restrict__ y,
                                int Nvox, int C, int doRelu) {
  int i = blockIdx.x * blockDim.x + threadIdx.x;
  if (i >= Nvox * C) return;
  int c = i % C;
  float t = (x[i] - stats[2 * C + c]) * stats[3 * C + c] * g[c] + b[c];
  if (res) t += (float)res[i];
  if (doRelu) t = fmaxf(t, 0.f);
  y[i] = (_Float16)t;
}

__global__ void copy_ch_kernel(const _Float16* __restrict__ src, _Float16* __restrict__ dst,
                               int Nvox, int Cs, int Cd, int cOff) {
  int i = blockIdx.x * blockDim.x + threadIdx.x;
  if (i >= Nvox * Cs) return;
  int v = i / Cs, c = i - v * Cs;
  dst[v * Cd + cOff + c] = src[i];
}

__global__ void gather_head_kernel(const float* __restrict__ pcd, const _Float16* __restrict__ b8,
                                   const float* __restrict__ wf, const float* __restrict__ bf,
                                   float* __restrict__ out, int npts) {
  int i = blockIdx.x * blockDim.x + threadIdx.x;
  if (i >= npts * 64) return;
  int n = i >> 6, f = i & 63;
  int ix = (int)floorf(pcd[n * 4 + 0] / 0.1f);
  int iy = (int)floorf(pcd[n * 4 + 1] / 0.1f);
  int iz = (int)floorf(pcd[n * 4 + 2] / 0.1f);
  int it = (int)floorf(pcd[n * 4 + 3] / 1.0f);
  ix = ix < 0 ? 0 : (ix > 31 ? 31 : ix);
  iy = iy < 0 ? 0 : (iy > 31 ? 31 : iy);
  iz = iz < 0 ? 0 : (iz > 31 ? 31 : iz);
  it = it < 0 ? 0 : (it > 3  ? 3  : it);
  int v = ((ix * 32 + iy) * 32 + iz) * 4 + it;
  float s = bf[f];
#pragma unroll
  for (int c = 0; c < 8; ++c) s += (float)b8[v * 8 + c] * wf[f * 8 + c];
  out[i] = s;
}

// ---------------------------------------------------------------------------
// Host orchestration
// ---------------------------------------------------------------------------
template <int SI, int SO, int CIN, int COUT, int K, int STRIDE, int PX, int PT, int MODE>
static void conv_t(const _Float16* in, const float* w, float* out, _Float16* wpack, hipStream_t s) {
  constexpr int TAPS   = K * K * K * K;
  constexpr int KTOT   = TAPS * CIN;
  constexpr int NCHUNK = (KTOT + 31) / 32;
  constexpr int TILESN = (COUT + 15) / 16;
  constexpr int NT     = (TILESN >= 4) ? 4 : TILESN;
  constexpr int NGROUP = TILESN / NT;
  constexpr int NVOX   = SO * SO * SO * 4;
  constexpr int TILESM = NVOX / 16;
  constexpr bool STAGE = (TILESM % 8) == 0;

  PackDesc pd; pd.Cin = CIN; pd.Cout = COUT; pd.taps = TAPS;
  const int packThreads = NCHUNK * TILESN * 32;
  pack_w_kernel<<<(packThreads + 255) / 256, 256, 0, s>>>(w, wpack, pd, NCHUNK);

  const int waves  = TILESM * NGROUP;
  const int blocks = STAGE ? (waves / 8) : ((waves + 7) / 8);
  conv_wmma_t<SI, SO, CIN, COUT, K, STRIDE, PX, PT, MODE, NT>
      <<<blocks, 256, 0, s>>>(in, wpack, out);
}

static void run_bn(const float* x, float* stats, const float* g, const float* b,
                   const _Float16* res, _Float16* y, int Nv, int C, bool relu, hipStream_t s) {
  zero_kernel<<<(2 * C + 255) / 256, 256, 0, s>>>(stats, 2 * C);
  int slabs = (Nv + 255) / 256; if (slabs > 64) slabs = 64; if (slabs < 1) slabs = 1;
  bn_stats_kernel<<<dim3(C, slabs), 256, 0, s>>>(x, stats, Nv, C);
  bn_finalize_kernel<<<(C + 255) / 256, 256, 0, s>>>(stats, Nv, C);
  int n = Nv * C;
  bn_apply_kernel<<<(n + 255) / 256, 256, 0, s>>>(x, stats, g, b, res, y, Nv, C, relu ? 1 : 0);
}

struct BP { const float *w1, *g1, *b1, *w2, *g2, *b2, *wd, *gd, *bd; };

template <int S, int CIN, int COUT>
static void run_block_t(const _Float16* x, _Float16* out, _Float16* t1h, _Float16* trh,
                        float* cf1, float* cf2, float* stats, const BP& p,
                        _Float16* wpack, hipStream_t s) {
  constexpr int NV = S * S * S * 4;
  conv_t<S, S, CIN, COUT, 3, 1, 1, 1, 0>(x, p.w1, cf1, wpack, s);
  run_bn(cf1, stats, p.g1, p.b1, nullptr, t1h, NV, COUT, true, s);
  conv_t<S, S, COUT, COUT, 3, 1, 1, 1, 0>(t1h, p.w2, cf2, wpack, s);
  const _Float16* res = x;   // valid only when CIN==COUT (no-wd blocks)
  if (p.wd) {
    conv_t<S, S, CIN, COUT, 1, 1, 0, 0, 0>(x, p.wd, cf1, wpack, s);
    run_bn(cf1, stats, p.gd, p.bd, nullptr, trh, NV, COUT, false, s);
    res = trh;
  }
  run_bn(cf2, stats, p.g2, p.b2, res, out, NV, COUT, true, s);
}

// Input flattening: jax tree order (sorted dict keys, depth-first).
enum {
  I_B0 = 0,
  I_B1BASE = 1,   // b1_: b1,b2,g1,g2,w1,w2 (no downsample)
  I_B2BASE = 7,   // b2_..b8_: b1,b2,bd,g1,g2,gd,w1,w2,wd (9 each)
  I_B3BASE = 16, I_B4BASE = 25, I_B5BASE = 34, I_B6BASE = 43, I_B7BASE = 52, I_B8BASE = 61,
  I_BC1 = 70, I_BC2 = 71, I_BC3 = 72, I_BC4 = 73,
  I_BF = 74,
  I_BT4 = 75, I_BT5 = 76, I_BT6 = 77, I_BT7 = 78,
  I_G0 = 79,
  I_GC1 = 80, I_GC2 = 81, I_GC3 = 82, I_GC4 = 83,
  I_GT4 = 84, I_GT5 = 85, I_GT6 = 86, I_GT7 = 87,
  I_W0 = 88,
  I_WC1 = 89, I_WC2 = 90, I_WC3 = 91, I_WC4 = 92,
  I_WF = 93,
  I_WT4 = 94, I_WT5 = 95, I_WT6 = 96, I_WT7 = 97,
  I_PCD = 98
};

extern "C" void kernel_launch(void* const* d_in, const int* in_sizes, int n_in,
                              void* d_out, int out_size, void* d_ws, size_t ws_size,
                              hipStream_t stream) {
  (void)n_in; (void)out_size; (void)ws_size;
  auto F = [&](int i) { return (const float*)d_in[i]; };

  auto mkbp_nd = [&](int base) {
    BP p; p.b1 = F(base + 0); p.b2 = F(base + 1); p.g1 = F(base + 2); p.g2 = F(base + 3);
    p.w1 = F(base + 4); p.w2 = F(base + 5); p.wd = nullptr; p.gd = nullptr; p.bd = nullptr;
    return p;
  };
  auto mkbp = [&](int base) {
    BP p; p.b1 = F(base + 0); p.b2 = F(base + 1); p.bd = F(base + 2);
    p.g1 = F(base + 3); p.g2 = F(base + 4); p.gd = F(base + 5);
    p.w1 = F(base + 6); p.w2 = F(base + 7); p.wd = F(base + 8);
    return p;
  };

  const float* pcd = F(I_PCD);
  const int Npts = in_sizes[I_PCD] / 4;

  const int N0 = 32 * 32 * 32 * 4;
  const int N1 = 16 * 16 * 16 * 4;
  const int N2 = 8 * 8 * 8 * 4;
  const int N3 = 4 * 4 * 4 * 4;

  // byte-based bump allocator (256B aligned)
  char* base = (char*)d_ws;
  size_t off = 0;
  auto allocB = [&](size_t bytes) { void* p = base + off; off = (off + bytes + 255) & ~(size_t)255; return p; };
  auto allocF = [&](size_t n) { return (float*)allocB(n * 4); };
  auto allocH = [&](size_t n) { return (_Float16*)allocB(n * 2); };

  float*     stats = allocF(4 * 384);
  _Float16*  wpack = allocH((size_t)8 * 1024 * 1024);   // >= 7.97M halves (b5.w1)
  float*     cf1   = allocF((size_t)N0 * 8);            // conv f32 scratch
  float*     cf2   = allocF((size_t)N0 * 8);

  _Float16* A0   = allocH((size_t)N0 * 1);
  _Float16* P1   = allocH((size_t)N0 * 8);
  _Float16* t1h  = allocH((size_t)N0 * 8);
  _Float16* trh  = allocH((size_t)N0 * 8);
  _Float16* c1   = allocH((size_t)N1 * 8);
  _Float16* B1   = allocH((size_t)N1 * 8);
  _Float16* c2   = allocH((size_t)N2 * 8);
  _Float16* B2   = allocH((size_t)N2 * 32);
  _Float16* c3   = allocH((size_t)N3 * 32);
  _Float16* B3   = allocH((size_t)N3 * 128);
  _Float16* c4   = allocH((size_t)32 * 128);
  _Float16* B4   = allocH((size_t)32 * 256);
  _Float16* up4  = allocH((size_t)N3 * 256);
  _Float16* cat5 = allocH((size_t)N3 * 384);
  _Float16* B5   = allocH((size_t)N3 * 256);
  _Float16* up5  = allocH((size_t)N2 * 128);
  _Float16* cat6 = allocH((size_t)N2 * 160);
  _Float16* B6   = allocH((size_t)N2 * 128);
  _Float16* up6  = allocH((size_t)N1 * 32);
  _Float16* cat7 = allocH((size_t)N1 * 40);
  _Float16* B7   = allocH((size_t)N1 * 32);
  _Float16* up7  = allocH((size_t)N0 * 8);
  _Float16* cat8 = allocH((size_t)N0 * 16);
  _Float16* B8   = allocH((size_t)N0 * 8);

  // ---- voxelize ----
  zero_h_kernel<<<(N0 + 255) / 256, 256, 0, stream>>>(A0, N0);
  voxelize_kernel<<<(Npts + 255) / 256, 256, 0, stream>>>(pcd, A0, Npts);

  // ---- encoder ----
  conv_t<32, 32, 1, 8, 5, 1, 2, 2, 0>(A0, F(I_W0), cf1, wpack, stream);
  run_bn(cf1, stats, F(I_G0), F(I_B0), nullptr, P1, N0, 8, true, stream);

  conv_t<32, 16, 8, 8, 2, 2, 0, 0, 0>(P1, F(I_WC1), cf1, wpack, stream);
  run_bn(cf1, stats, F(I_GC1), F(I_BC1), nullptr, c1, N1, 8, true, stream);
  run_block_t<16, 8, 8>(c1, B1, t1h, trh, cf1, cf2, stats, mkbp_nd(I_B1BASE), wpack, stream);

  conv_t<16, 8, 8, 8, 2, 2, 0, 0, 0>(B1, F(I_WC2), cf1, wpack, stream);
  run_bn(cf1, stats, F(I_GC2), F(I_BC2), nullptr, c2, N2, 8, true, stream);
  run_block_t<8, 8, 32>(c2, B2, t1h, trh, cf1, cf2, stats, mkbp(I_B2BASE), wpack, stream);

  conv_t<8, 4, 32, 32, 2, 2, 0, 0, 0>(B2, F(I_WC3), cf1, wpack, stream);
  run_bn(cf1, stats, F(I_GC3), F(I_BC3), nullptr, c3, N3, 32, true, stream);
  run_block_t<4, 32, 128>(c3, B3, t1h, trh, cf1, cf2, stats, mkbp(I_B3BASE), wpack, stream);

  conv_t<4, 2, 128, 128, 2, 2, 0, 0, 0>(B3, F(I_WC4), cf1, wpack, stream);
  run_bn(cf1, stats, F(I_GC4), F(I_BC4), nullptr, c4, 32, 128, true, stream);
  run_block_t<2, 128, 256>(c4, B4, t1h, trh, cf1, cf2, stats, mkbp(I_B4BASE), wpack, stream);

  // ---- decoder ----
  conv_t<2, 4, 256, 256, 2, 1, 0, 0, 1>(B4, F(I_WT4), cf1, wpack, stream);
  run_bn(cf1, stats, F(I_GT4), F(I_BT4), nullptr, up4, N3, 256, true, stream);
  copy_ch_kernel<<<(N3 * 256 + 255) / 256, 256, 0, stream>>>(up4, cat5, N3, 256, 384, 0);
  copy_ch_kernel<<<(N3 * 128 + 255) / 256, 256, 0, stream>>>(B3, cat5, N3, 128, 384, 256);
  run_block_t<4, 384, 256>(cat5, B5, t1h, trh, cf1, cf2, stats, mkbp(I_B5BASE), wpack, stream);

  conv_t<4, 8, 256, 128, 2, 1, 0, 0, 1>(B5, F(I_WT5), cf1, wpack, stream);
  run_bn(cf1, stats, F(I_GT5), F(I_BT5), nullptr, up5, N2, 128, true, stream);
  copy_ch_kernel<<<(N2 * 128 + 255) / 256, 256, 0, stream>>>(up5, cat6, N2, 128, 160, 0);
  copy_ch_kernel<<<(N2 * 32 + 255) / 256, 256, 0, stream>>>(B2, cat6, N2, 32, 160, 128);
  run_block_t<8, 160, 128>(cat6, B6, t1h, trh, cf1, cf2, stats, mkbp(I_B6BASE), wpack, stream);

  conv_t<8, 16, 128, 32, 2, 1, 0, 0, 1>(B6, F(I_WT6), cf1, wpack, stream);
  run_bn(cf1, stats, F(I_GT6), F(I_BT6), nullptr, up6, N1, 32, true, stream);
  copy_ch_kernel<<<(N1 * 32 + 255) / 256, 256, 0, stream>>>(up6, cat7, N1, 32, 40, 0);
  copy_ch_kernel<<<(N1 * 8 + 255) / 256, 256, 0, stream>>>(B1, cat7, N1, 8, 40, 32);
  run_block_t<16, 40, 32>(cat7, B7, t1h, trh, cf1, cf2, stats, mkbp(I_B7BASE), wpack, stream);

  conv_t<16, 32, 32, 8, 2, 1, 0, 0, 1>(B7, F(I_WT7), cf1, wpack, stream);
  run_bn(cf1, stats, F(I_GT7), F(I_BT7), nullptr, up7, N0, 8, true, stream);
  copy_ch_kernel<<<(N0 * 8 + 255) / 256, 256, 0, stream>>>(up7, cat8, N0, 8, 16, 0);
  copy_ch_kernel<<<(N0 * 8 + 255) / 256, 256, 0, stream>>>(P1, cat8, N0, 8, 16, 8);
  run_block_t<32, 16, 8>(cat8, B8, t1h, trh, cf1, cf2, stats, mkbp(I_B8BASE), wpack, stream);

  // ---- fused final pointwise (8 -> 64) + bias + gather ----
  gather_head_kernel<<<(Npts * 64 + 255) / 256, 256, 0, stream>>>(
      pcd, B8, F(I_WF), F(I_BF), (float*)d_out, Npts);
}
